// _ProposalLayer_67628555043302
// MI455X (gfx1250) — matrix operand order repaired and checked by reading, not compiled, not run on hardware
//
#include <hip/hip_runtime.h>
#include <math.h>

typedef __attribute__((ext_vector_type(16))) _Float16 v16h;
typedef __attribute__((ext_vector_type(8)))  float    v8f;
typedef unsigned int u32x4 __attribute__((ext_vector_type(4)));
typedef int          i32x8 __attribute__((ext_vector_type(8)));
typedef int          i32x4 __attribute__((ext_vector_type(4)));

#define BATCH    32
#define NANCH    15
#define HW       9216          /* 96*96 */
#define NPER     (NANCH * HW)  /* 138240 */
#define PRE_NMS  1024
#define POST_NMS 256
#define SEARCH_M1 767.0f
#define NMS_THR  0.7f

// ---------------------------------------------------------------------------
// Kernel 1: per-element 2-way log-softmax -> class-1 score
// ---------------------------------------------------------------------------
__global__ void score_kernel(const float* __restrict__ cls,
                             float* __restrict__ scores) {
  int g = blockIdx.x * blockDim.x + threadIdx.x;
  if (g >= BATCH * NPER) return;
  int b = g / NPER, i = g - b * NPER;
  int a = i / HW, hw = i - a * HW;
  const float* base = cls + ((size_t)b * 30 + a) * HW + hw;
  float c0 = base[0];
  float c1 = base[(size_t)NANCH * HW];
  float m  = fmaxf(c0, c1);
  float lse = m + __logf(__expf(c0 - m) + __expf(c1 - m));
  scores[g] = c1 - lse;
}

// ---------------------------------------------------------------------------
// Kernel 2: per-batch exact top-1024 via 4-pass radix select + bitonic sort
// One 1024-thread block (32 waves) per batch.
// ---------------------------------------------------------------------------
__global__ void __launch_bounds__(1024)
topk_kernel(const float* __restrict__ scores,
            float* __restrict__ sel_score,
            int* __restrict__ sel_idx) {
  __shared__ unsigned hist[256];
  __shared__ unsigned sh_prefix;
  __shared__ int sh_k;
  __shared__ int sh_cntGT, sh_cntEQ;
  __shared__ unsigned keyS[PRE_NMS];
  __shared__ int idxS[PRE_NMS];

  int b = blockIdx.x;
  int tid = threadIdx.x;
  const float* sc = scores + (size_t)b * NPER;

  if (tid == 0) { sh_prefix = 0u; sh_k = PRE_NMS; }

  for (int bp = 24; bp >= 0; bp -= 8) {
    if (tid < 256) hist[tid] = 0u;
    __syncthreads();
    unsigned prefix = sh_prefix;
    unsigned mask = (bp == 24) ? 0u : (0xFFFFFFFFu << (bp + 8));
    for (int i = tid; i < NPER; i += 1024) {
      unsigned u = __float_as_uint(sc[i]);
      unsigned key = u ^ ((u & 0x80000000u) ? 0xFFFFFFFFu : 0x80000000u);
      if ((key & mask) == prefix)
        atomicAdd(&hist[(key >> bp) & 255u], 1u);
    }
    __syncthreads();
    if (tid == 0) {
      int k = sh_k;
      int cum = 0;
      int sel = 0;
      for (int bin = 255; bin >= 0; --bin) {
        int h = (int)hist[bin];
        if (cum + h >= k) { sel = bin; sh_k = k - cum; break; }
        cum += h;
      }
      sh_prefix = prefix | ((unsigned)sel << bp);
    }
    __syncthreads();
  }

  unsigned T = sh_prefix;       // exact 1024th-largest key
  int kEq = sh_k;               // how many ==T to take
  int numGT = PRE_NMS - kEq;
  if (tid == 0) { sh_cntGT = 0; sh_cntEQ = 0; }
  __syncthreads();

  for (int i = tid; i < NPER; i += 1024) {
    unsigned u = __float_as_uint(sc[i]);
    unsigned key = u ^ ((u & 0x80000000u) ? 0xFFFFFFFFu : 0x80000000u);
    if (key > T) {
      int p = atomicAdd(&sh_cntGT, 1);
      keyS[p] = key; idxS[p] = i;
    } else if (key == T) {
      int e = atomicAdd(&sh_cntEQ, 1);
      if (e < kEq) { keyS[numGT + e] = key; idxS[numGT + e] = i; }
    }
  }
  __syncthreads();

  // bitonic sort, descending by key
  for (int k = 2; k <= PRE_NMS; k <<= 1) {
    for (int j = k >> 1; j > 0; j >>= 1) {
      int ixj = tid ^ j;
      if (ixj > tid) {
        bool descSeg = ((tid & k) == 0);
        unsigned ka = keyS[tid], kb = keyS[ixj];
        if ((ka < kb) == descSeg) {
          keyS[tid] = kb; keyS[ixj] = ka;
          int t = idxS[tid]; idxS[tid] = idxS[ixj]; idxS[ixj] = t;
        }
      }
      __syncthreads();
    }
  }

  unsigned key = keyS[tid];
  unsigned u = (key & 0x80000000u) ? (key ^ 0x80000000u) : ~key;
  sel_score[b * PRE_NMS + tid] = __uint_as_float(u);
  sel_idx[b * PRE_NMS + tid] = idxS[tid];
}

// ---------------------------------------------------------------------------
// Kernel 3: decode the 1024 selected boxes per batch (anchors recomputed)
// ---------------------------------------------------------------------------
__global__ void decode_kernel(const float* __restrict__ reg,
                              const int* __restrict__ sel_idx,
                              float* __restrict__ boxes) {
  int g = blockIdx.x * blockDim.x + threadIdx.x;
  if (g >= BATCH * PRE_NMS) return;
  int b = g >> 10;
  int idx = sel_idx[g];
  int a = idx / HW, hw = idx - a * HW;
  int row = hw / 96, col = hw - row * 96;
  const float ratios[5] = {0.33f, 0.5f, 1.0f, 2.0f, 3.0f};
  int ri = a / 3, si = a - ri * 3;
  float scale = (float)(4 << si);        // 4, 8, 16
  float sq = sqrtf(ratios[ri]);
  float w = 8.0f * scale / sq;
  float h = 8.0f * scale * sq;
  float aw = w + 1.0f, ah = h + 1.0f;
  float acx = 8.0f * (float)col + 0.5f;  // ori == 0
  float acy = 8.0f * (float)row + 0.5f;
  const float* rb = reg + ((size_t)b * 60 + a) * HW + hw;
  float dx = rb[0];
  float dy = rb[(size_t)NANCH * HW];
  float dw = rb[(size_t)2 * NANCH * HW];
  float dh = rb[(size_t)3 * NANCH * HW];
  float pcx = dx * aw + acx;
  float pcy = dy * ah + acy;
  float pw = __expf(dw) * aw;
  float ph = __expf(dh) * ah;
  float x1 = fminf(fmaxf(pcx - 0.5f * pw, 0.0f), SEARCH_M1);
  float y1 = fminf(fmaxf(pcy - 0.5f * ph, 0.0f), SEARCH_M1);
  float x2 = fminf(fmaxf(pcx + 0.5f * pw, 0.0f), SEARCH_M1);
  float y2 = fminf(fmaxf(pcy + 0.5f * ph, 0.0f), SEARCH_M1);
  float4 bx = {x1, y1, x2, y2};
  ((float4*)boxes)[g] = bx;
}

// ---------------------------------------------------------------------------
// Kernel 4: TDM-staged greedy NMS + WMMA prefix-sum compaction + output
// One 1024-thread block per batch.
// ---------------------------------------------------------------------------
__global__ void __launch_bounds__(1024)
nms_kernel(const float* __restrict__ boxes,
           const float* __restrict__ sel_score,
           float* __restrict__ out) {
  __shared__ float4 sbox[PRE_NMS];   // 16 KB tile, staged by the TDM
  __shared__ float sar[PRE_NMS];
  __shared__ int   keep[PRE_NMS];
  __shared__ float flg[PRE_NMS];
  __shared__ float scanp[PRE_NMS];   // within-row inclusive prefixes (WMMA out)
  __shared__ float rowsum[32];
  __shared__ float rowcarry[32];

  int b = blockIdx.x;
  int tid = threadIdx.x;

  // ---- Tensor Data Mover: DMA this batch's 1024 boxes (16 KB) into LDS ----
  // D# group0: count=1 | lds_addr | global_addr[56:0] | type=2
  // D# group1: data_size=4B, tensor_dim0=tile_dim0=4096 elems (1 row),
  //            tensor_dim1=tile_dim1=1, dim0_stride=4096
  if (tid < 32) {   // one wave issues; TENSORcnt tracked per issuing wave
    unsigned long long ga =
        (unsigned long long)(boxes + (size_t)b * PRE_NMS * 4);
    unsigned lds = (unsigned)(unsigned long long)&sbox[0]; // flat[31:0] = LDS offset
    u32x4 g0 = { 1u,                                   // count=1 (valid, user)
                 lds,                                  // lds_addr [63:32]
                 (unsigned)(ga & 0xFFFFFFFFull),       // global_addr lo
                 (unsigned)((ga >> 32) & 0x01FFFFFFull) | 0x80000000u }; // type=2
    i32x8 g1 = { (int)0x00020000,                      // data_size=2 (4 bytes)
                 (int)(4096u << 16),                   // tensor_dim0[15:0] @ [63:48]
                 (int)(1u << 16),                      // tensor_dim1[15:0] @ [95:80]
                 (int)(4096u << 16),                   // tile_dim0 @ [127:112]
                 1,                                    // tile_dim1 @ [143:128]
                 4096,                                 // tensor_dim0_stride lo
                 0, 0 };
    i32x4 gz4 = { 0, 0, 0, 0 };                        // <=2D tensor: groups 2/3 null
    i32x8 gz8 = { 0, 0, 0, 0, 0, 0, 0, 0 };
    __builtin_amdgcn_tensor_load_to_lds(g0, g1, gz4, gz4, gz8, 0);
    __builtin_amdgcn_s_wait_tensorcnt(0);
  }
  __syncthreads();

  float4 bx = sbox[tid];
  float mya = (bx.z - bx.x + 1.0f) * (bx.w - bx.y + 1.0f);
  sar[tid] = mya;
  keep[tid] = 1;
  // zero this batch's 1280-float output slab (harness poisons, we must clear)
  out[(size_t)b * (POST_NMS * 5) + tid] = 0.0f;
  if (tid < POST_NMS * 5 - PRE_NMS)
    out[(size_t)b * (POST_NMS * 5) + PRE_NMS + tid] = 0.0f;
  __syncthreads();

  // greedy NMS: serial over i, parallel over j (boxes pre-sorted by score)
  for (int i = 0; i < PRE_NMS - 1; ++i) {
    if (keep[i] && tid > i) {
      float4 bi = sbox[i];                 // LDS broadcast read
      float ix1 = fmaxf(bi.x, bx.x);
      float iy1 = fmaxf(bi.y, bx.y);
      float ix2 = fminf(bi.z, bx.z);
      float iy2 = fminf(bi.w, bx.w);
      float iw = fmaxf(ix2 - ix1 + 1.0f, 0.0f);
      float ih = fmaxf(iy2 - iy1 + 1.0f, 0.0f);
      float inter = iw * ih;
      float iou = inter / (sar[i] + mya - inter);
      if (iou > NMS_THR) keep[tid] = 0;
    }
    __syncthreads();
  }

  flg[tid] = keep[tid] ? 1.0f : 0.0f;
  __syncthreads();

  // -------- prefix sum of the 1024 keep-flags on the WMMA unit (wave 0) ----
  // Two 16x32 f16 tiles X;  D1 = X*U1 (prefix cols 0..15), D2 = X*U2
  // (prefix cols 16..31), P = X*J (row sums).  0/1 flags are exact in f16.
  if (tid < 32) {
    int lane = tid;
    int n = lane & 15;
    int rbase = (lane < 16) ? 0 : 8;

    // B matrices (32x16): VGPRs 0-3 -> K 0..7 (+8 for hi lanes), VGPRs 4-7 -> K 16..23 (+8 for hi lanes)
    v16h u1, u2, onesB;
    for (int r = 0; r < 8; ++r) {
      for (int s = 0; s < 2; ++s) {
        int kg = ((r < 4) ? 0 : 16) + ((lane < 16) ? 0 : 8) + 2 * (r & 3) + s;
        u1[2 * r + s] = (kg <= n)      ? (_Float16)1.0f : (_Float16)0.0f;
        u2[2 * r + s] = (kg <= n + 16) ? (_Float16)1.0f : (_Float16)0.0f;
      }
    }
    for (int t = 0; t < 16; ++t) onesB[t] = (_Float16)1.0f;

    for (int c = 0; c < 2; ++c) {
      // A matrix (16x32): lane = row; VGPRs 0-3 -> K 0..7 (+8 for hi lanes),
      // VGPRs 4-7 -> K 16..23 (+8 for hi lanes)
      v16h aM;
      int rowA = lane & 15;
      for (int v = 0; v < 8; ++v) {
        int kb = ((v < 4) ? 0 : 16) + ((lane < 16) ? 0 : 8) + 2 * (v & 3);
        aM[2 * v]     = (_Float16)flg[c * 512 + rowA * 32 + kb];
        aM[2 * v + 1] = (_Float16)flg[c * 512 + rowA * 32 + kb + 1];
      }
      v8f z = {};
      v8f d1 = __builtin_amdgcn_wmma_f32_16x16x32_f16(false, aM, false, u1,
                                                      (short)0, z, false, false);
      v8f d2 = __builtin_amdgcn_wmma_f32_16x16x32_f16(false, aM, false, u2,
                                                      (short)0, z, false, false);
      v8f p  = __builtin_amdgcn_wmma_f32_16x16x32_f16(false, aM, false, onesB,
                                                      (short)0, z, false, false);
      // D layout: VGPR v -> row v (lanes 0-15) / v+8 (lanes 16-31), col = n
      for (int v = 0; v < 8; ++v) {
        int row = v + rbase;
        scanp[c * 512 + row * 32 + n]      = d1[v];
        scanp[c * 512 + row * 32 + 16 + n] = d2[v];
        if (n == 0) rowsum[c * 16 + row] = p[v];
      }
    }
    if (lane == 0) {           // tiny serial carry scan over 32 row sums
      float acc = 0.0f;
      for (int r = 0; r < 32; ++r) { rowcarry[r] = acc; acc += rowsum[r]; }
    }
  }
  __syncthreads();

  // inclusive scan value = within-row prefix + carry of preceding rows
  int pos = (int)(scanp[tid] + rowcarry[tid >> 5]) - 1;

  if (keep[tid] && pos < POST_NMS) {
    float* o = out + (size_t)b * (POST_NMS * 5) + (size_t)pos * 5;
    o[0] = sel_score[b * PRE_NMS + tid];
    o[1] = bx.x; o[2] = bx.y; o[3] = bx.z; o[4] = bx.w;
  }
}

// ---------------------------------------------------------------------------
extern "C" void kernel_launch(void* const* d_in, const int* in_sizes, int n_in,
                              void* d_out, int out_size, void* d_ws, size_t ws_size,
                              hipStream_t stream) {
  const float* pred_cls = (const float*)d_in[0];
  const float* pred_reg = (const float*)d_in[1];
  // d_in[2] (anchors) intentionally unused: recomputed analytically on-device
  float* out = (float*)d_out;

  float* ws_scores = (float*)d_ws;                                  // B*N
  int*   sel_idx   = (int*)(ws_scores + (size_t)BATCH * NPER);      // B*1024
  float* sel_score = (float*)(sel_idx + BATCH * PRE_NMS);           // B*1024
  float* boxes     = sel_score + BATCH * PRE_NMS;                   // B*1024*4

  score_kernel<<<(BATCH * NPER + 255) / 256, 256, 0, stream>>>(pred_cls, ws_scores);
  topk_kernel<<<BATCH, 1024, 0, stream>>>(ws_scores, sel_score, sel_idx);
  decode_kernel<<<(BATCH * PRE_NMS + 255) / 256, 256, 0, stream>>>(pred_reg, sel_idx, boxes);
  nms_kernel<<<BATCH, 1024, 0, stream>>>(boxes, sel_score, out);
}
